// Sboxblock_90623809946144
// MI455X (gfx1250) — matrix operand order repaired and checked by reading, not compiled, not run on hardware
//
#include <hip/hip_runtime.h>
#include <hip/hip_bf16.h>

// ---------------------------------------------------------------------------
// 3D Swin block for MI455X (gfx1250, wave32).  All GEMMs via
// v_wmma_f32_16x16x32_f16.  B=4, R=32, C=96, W=4, HEADS=4, dh=24 (pad 32),
// N=64 tokens/window, NW=512 windows/batch, MLP=384.
//
// All WMMA B-operands are stored K-contiguous (transposed weights / vT) so
// every fragment load is two 16-byte vector loads per lane.
// ---------------------------------------------------------------------------

typedef __attribute__((ext_vector_type(16))) _Float16 v16h;
typedef __attribute__((ext_vector_type(8)))  _Float16 h8;
typedef __attribute__((ext_vector_type(8)))  float    v8f;

#define WMMA16(a, b, c) \
  __builtin_amdgcn_wmma_f32_16x16x32_f16(false, (a), false, (b), (short)0, (c), false, false)

// ---- constants -------------------------------------------------------------
#define CB        96          // channels
#define C3        288         // 3*C
#define MLPD      384
#define NTOK      131072      // B * R^3 rows (also box rows: 2048 windows * 64)
#define DPAD      32          // padded head dim
#define QSZ       16777216    // halves per q/k/vT buffer

// ---- WMMA operand loaders (CDNA5 wave32 layouts, ISA 7.12.2) ---------------

// A: 16x32 f16.  lane: row = row0 + (l&15); halves e: k = (e&7)+(e>>3)*16+(l>>4)*8
static __device__ __forceinline__ v16h load_a16x32(const _Float16* A, int lda,
                                                   int row0, int k0) {
  const int lane = threadIdx.x & 31;
  const _Float16* p = A + (long)(row0 + (lane & 15)) * lda + k0 + ((lane >> 4) * 8);
  h8 lo = *(const h8*)(p);
  h8 hi = *(const h8*)(p + 16);
  v16h a;
#pragma unroll
  for (int i = 0; i < 8; ++i) { a[i] = lo[i]; a[i + 8] = hi[i]; }
  return a;
}

// B: 32x16, stored K-contiguous: logical B[k][n] = base[n*ldk + k].
// lane: col = col0 + (l&15); halves e: k = k0 + (e&7)+(e>>3)*16+(l>>4)*8
static __device__ __forceinline__ v16h load_bT(const _Float16* base, int ldk,
                                               int k0, int col0) {
  const int lane = threadIdx.x & 31;
  const _Float16* p = base + (long)(col0 + (lane & 15)) * ldk + k0 + ((lane >> 4) * 8);
  h8 lo = *(const h8*)(p);
  h8 hi = *(const h8*)(p + 16);
  v16h b;
#pragma unroll
  for (int i = 0; i < 8; ++i) { b[i] = lo[i]; b[i + 8] = hi[i]; }
  return b;
}

// multi-N-tile GEMM accumulate: one A fragment reused across NT B tiles
template <int NT, int KC>
static __device__ __forceinline__ void gemm_acc(const _Float16* A, int lda,
                                                const _Float16* Bt, int ldk,
                                                int mt, int ng, v8f acc[NT]) {
#pragma unroll
  for (int t = 0; t < NT; ++t)
#pragma unroll
    for (int i = 0; i < 8; ++i) acc[t][i] = 0.0f;
#pragma unroll
  for (int kc = 0; kc < KC; ++kc) {
    const v16h a = load_a16x32(A, lda, mt * 16, kc * 32);
#pragma unroll
    for (int t = 0; t < NT; ++t)
      acc[t] = WMMA16(a, load_bT(Bt, ldk, kc * 32, (ng * NT + t) * 16), acc[t]);
  }
}

// box row (0..131071) -> token index within batch (applies roll by SHIFT=2)
static __device__ __forceinline__ int box_to_token(int m) {
  const int wa = m >> 6, n = m & 63;
  const int widx = wa & 511;
  const int bx = widx >> 6, by = (widx >> 3) & 7, bz = widx & 7;
  const int ix = n >> 4, iy = (n >> 2) & 3, iz = n & 3;
  const int x = (bx * 4 + ix + 2) & 31;
  const int y = (by * 4 + iy + 2) & 31;
  const int z = (bz * 4 + iz + 2) & 31;
  return (x * 32 + y) * 32 + z;
}

// ---- small utility kernels -------------------------------------------------

// f32 (K x N) -> f16 transposed (N x K): Wt[n*K + k] = W[k*N + n]
__global__ void k_w_to_f16t(const float* __restrict__ src,
                            _Float16* __restrict__ dst, int K, int N) {
  int i = blockIdx.x * blockDim.x + threadIdx.x;
  if (i < K * N) {
    int k = i / N, n = i - k * N;
    dst[n * K + k] = (_Float16)src[i];
  }
}

__global__ void k_zero128(uint4* __restrict__ dst, int n) {
  int i = blockIdx.x * blockDim.x + threadIdx.x;
  if (i < n) dst[i] = make_uint4(0u, 0u, 0u, 0u);
}

// ---- LN1 + roll + box partition -> x_boxed (f16) ---------------------------
__global__ void k_ln1_box(const float* __restrict__ in, const float* __restrict__ g,
                          const float* __restrict__ be, _Float16* __restrict__ xbox) {
  const int m = blockIdx.x * (blockDim.x >> 5) + (threadIdx.x >> 5);
  const int lane = threadIdx.x & 31;
  const int b = m >> 15;
  const int tok = box_to_token(m);
  const float* src = in + (long)((b << 15) | tok) * CB;
  float x0 = src[lane], x1 = src[lane + 32], x2 = src[lane + 64];
  float s = x0 + x1 + x2;
#pragma unroll
  for (int o = 16; o; o >>= 1) s += __shfl_xor(s, o, 32);
  const float mu = s * (1.0f / 96.0f);
  float d0 = x0 - mu, d1 = x1 - mu, d2 = x2 - mu;
  float vs = d0 * d0 + d1 * d1 + d2 * d2;
#pragma unroll
  for (int o = 16; o; o >>= 1) vs += __shfl_xor(vs, o, 32);
  const float inv = rsqrtf(vs * (1.0f / 96.0f) + 1e-5f);
  _Float16* dst = xbox + (long)m * CB;
  dst[lane]      = (_Float16)(d0 * inv * g[lane]      + be[lane]);
  dst[lane + 32] = (_Float16)(d1 * inv * g[lane + 32] + be[lane + 32]);
  dst[lane + 64] = (_Float16)(d2 * inv * g[lane + 64] + be[lane + 64]);
}

// ---- QKV GEMM: (131072x96)@(96x288)+bias -> q/k padded, vT -----------------
__global__ void __launch_bounds__(256) k_qkv(const _Float16* __restrict__ A,
                                             const _Float16* __restrict__ Wt,
                                             const float* __restrict__ bias,
                                             _Float16* __restrict__ q,
                                             _Float16* __restrict__ k,
                                             _Float16* __restrict__ vT) {
  const int wid = blockIdx.x * (blockDim.x >> 5) + (threadIdx.x >> 5);
  const int mt = wid / 6, ng = wid - mt * 6;          // 6 groups of 3 N-tiles
  v8f acc[3];
  gemm_acc<3, 3>(A, CB, Wt, CB, mt, ng, acc);
  const int lane = threadIdx.x & 31;
#pragma unroll
  for (int t = 0; t < 3; ++t) {
    const int col = (ng * 3 + t) * 16 + (lane & 15);
    const int s   = col / CB;
    const int rem = col - s * CB;
    const int hh  = rem / 24;
    const int d   = rem - hh * 24;
    const float scl = (s == 0) ? 0.20412414523193154f : 1.0f;  // 1/sqrt(24) on q
    const float bv  = bias[col];
#pragma unroll
    for (int vi = 0; vi < 8; ++vi) {
      const int row = mt * 16 + ((lane >> 4) * 8) + vi;
      const int w = row >> 6, n = row & 63;
      const float val = (acc[t][vi] + bv) * scl;
      if (s == 2) {
        vT[(((w * 4 + hh) * DPAD + d) << 6) + n] = (_Float16)val;   // [d][key]
      } else {
        _Float16* buf = (s == 0) ? q : k;
        buf[(((w * 4 + hh) * 64 + n) << 5) + d] = (_Float16)val;    // [key][d]
      }
    }
  }
}

// ---- fused window attention: one block per (window, head) ------------------
__global__ void __launch_bounds__(128) k_attn(const _Float16* __restrict__ q,
                                              const _Float16* __restrict__ k,
                                              const _Float16* __restrict__ vT,
                                              const float* __restrict__ mask,
                                              _Float16* __restrict__ out) {
  __shared__ __align__(16) float    S[64][68];
  __shared__ __align__(16) _Float16 P[64][64];
  const int wa = blockIdx.x >> 2;        // global window 0..2047
  const int h  = blockIdx.x & 3;         // head
  const int widx = wa & 511;             // window-in-batch (mask index)
  const long base = (long)(wa * 4 + h) << 11;   // *64*32
  const _Float16* qb = q  + base;
  const _Float16* kb = k  + base;
  const _Float16* vb = vT + base;
  const int wv = threadIdx.x >> 5, lane = threadIdx.x & 31;
  const int row0 = wv * 16;
  const float* mrow = mask + (long)widx * 4096;

  // S = q @ k^T  (K = 32 padded; scale already folded into q)
  const v16h aq = load_a16x32(qb, DPAD, row0, 0);
#pragma unroll
  for (int nt = 0; nt < 4; ++nt) {
    v8f acc;
#pragma unroll
    for (int i = 0; i < 8; ++i) acc[i] = 0.0f;
    acc = WMMA16(aq, load_bT(kb, DPAD, 0, nt * 16), acc);
    const int key = nt * 16 + (lane & 15);
#pragma unroll
    for (int vi = 0; vi < 8; ++vi) {
      const int mr = row0 + ((lane >> 4) * 8) + vi;
      S[mr][key] = acc[vi] + mrow[mr * 64 + key];
    }
  }

  // row-wise softmax on this wave's 16 rows (rows written by same wave: no bar)
  for (int r = 0; r < 16; ++r) {
    const int mr = row0 + r;
    float a0 = S[mr][lane], a1 = S[mr][lane + 32];
    float mx = fmaxf(a0, a1);
#pragma unroll
    for (int o = 16; o; o >>= 1) mx = fmaxf(mx, __shfl_xor(mx, o, 32));
    const float e0 = __expf(a0 - mx), e1 = __expf(a1 - mx);
    float sm = e0 + e1;
#pragma unroll
    for (int o = 16; o; o >>= 1) sm += __shfl_xor(sm, o, 32);
    const float inv = 1.0f / sm;
    P[mr][lane]      = (_Float16)(e0 * inv);
    P[mr][lane + 32] = (_Float16)(e1 * inv);
  }

  // out = P @ V   (K = 64 keys, 2 chunks; vT is [d][key] -> K-contiguous B)
#pragma unroll
  for (int nt = 0; nt < 2; ++nt) {
    v8f acc;
#pragma unroll
    for (int i = 0; i < 8; ++i) acc[i] = 0.0f;
#pragma unroll
    for (int kc = 0; kc < 2; ++kc)
      acc = WMMA16(load_a16x32(&P[0][0], 64, row0, kc * 32),
                   load_bT(vb, 64, kc * 32, nt * 16), acc);
    const int d = nt * 16 + (lane & 15);
    if (d < 24) {
#pragma unroll
      for (int vi = 0; vi < 8; ++vi) {
        const int mr = row0 + ((lane >> 4) * 8) + vi;
        out[(long)(wa * 64 + mr) * CB + h * 24 + d] = (_Float16)acc[vi];
      }
    }
  }
}

// ---- proj GEMM + box-reverse + roll + 0.5x + shortcut -> x1 (f32) ----------
__global__ void __launch_bounds__(256) k_proj(const _Float16* __restrict__ A,
                                              const _Float16* __restrict__ Wt,
                                              const float* __restrict__ bias,
                                              const float* __restrict__ shortcut,
                                              float* __restrict__ x1) {
  const int wid = blockIdx.x * (blockDim.x >> 5) + (threadIdx.x >> 5);
  const int mt = wid >> 1, ng = wid & 1;              // 2 groups of 3 N-tiles
  v8f acc[3];
  gemm_acc<3, 3>(A, CB, Wt, CB, mt, ng, acc);
  const int lane = threadIdx.x & 31;
#pragma unroll
  for (int t = 0; t < 3; ++t) {
    const int col = (ng * 3 + t) * 16 + (lane & 15);
    const float bv = bias[col];
#pragma unroll
    for (int vi = 0; vi < 8; ++vi) {
      const int row = mt * 16 + ((lane >> 4) * 8) + vi;
      const int b = row >> 15;
      const long idx = (long)((b << 15) | box_to_token(row)) * CB + col;
      x1[idx] = 0.5f * (acc[t][vi] + bv) + shortcut[idx];
    }
  }
}

// ---- LN2 (token order) -> h (f16) ------------------------------------------
__global__ void k_ln2(const float* __restrict__ x1, const float* __restrict__ g,
                      const float* __restrict__ be, _Float16* __restrict__ hbuf) {
  const int t = blockIdx.x * (blockDim.x >> 5) + (threadIdx.x >> 5);
  const int lane = threadIdx.x & 31;
  const float* src = x1 + (long)t * CB;
  float x0 = src[lane], x1v = src[lane + 32], x2 = src[lane + 64];
  float s = x0 + x1v + x2;
#pragma unroll
  for (int o = 16; o; o >>= 1) s += __shfl_xor(s, o, 32);
  const float mu = s * (1.0f / 96.0f);
  float d0 = x0 - mu, d1 = x1v - mu, d2 = x2 - mu;
  float vs = d0 * d0 + d1 * d1 + d2 * d2;
#pragma unroll
  for (int o = 16; o; o >>= 1) vs += __shfl_xor(vs, o, 32);
  const float inv = rsqrtf(vs * (1.0f / 96.0f) + 1e-5f);
  _Float16* dst = hbuf + (long)t * CB;
  dst[lane]      = (_Float16)(d0 * inv * g[lane]      + be[lane]);
  dst[lane + 32] = (_Float16)(d1 * inv * g[lane + 32] + be[lane + 32]);
  dst[lane + 64] = (_Float16)(d2 * inv * g[lane + 64] + be[lane + 64]);
}

static __device__ __forceinline__ float gelu_tanh(float x) {
  const float c = 0.7978845608028654f;        // sqrt(2/pi)
  const float t = tanhf(c * (x + 0.044715f * x * x * x));
  return 0.5f * x * (1.0f + t);
}

// ---- MLP1 GEMM + bias + GELU -> hidden (f16) -------------------------------
__global__ void __launch_bounds__(256) k_mlp1(const _Float16* __restrict__ A,
                                              const _Float16* __restrict__ Wt,
                                              const float* __restrict__ bias,
                                              _Float16* __restrict__ hid) {
  const int wid = blockIdx.x * (blockDim.x >> 5) + (threadIdx.x >> 5);
  const int mt = wid / 6, ng = wid - mt * 6;          // 6 groups of 4 N-tiles
  v8f acc[4];
  gemm_acc<4, 3>(A, CB, Wt, CB, mt, ng, acc);
  const int lane = threadIdx.x & 31;
#pragma unroll
  for (int t = 0; t < 4; ++t) {
    const int col = (ng * 4 + t) * 16 + (lane & 15);
    const float bv = bias[col];
#pragma unroll
    for (int vi = 0; vi < 8; ++vi) {
      const int row = mt * 16 + ((lane >> 4) * 8) + vi;
      hid[(long)row * MLPD + col] = (_Float16)gelu_tanh(acc[t][vi] + bv);
    }
  }
}

// ---- MLP2 GEMM + bias, out = 0.5*h + x1 -> d_out (f32) ---------------------
__global__ void __launch_bounds__(256) k_mlp2(const _Float16* __restrict__ A,
                                              const _Float16* __restrict__ Wt,
                                              const float* __restrict__ bias,
                                              const float* __restrict__ x1,
                                              float* __restrict__ outp) {
  const int wid = blockIdx.x * (blockDim.x >> 5) + (threadIdx.x >> 5);
  const int mt = wid >> 1, ng = wid & 1;              // 2 groups of 3 N-tiles
  v8f acc[3];
  gemm_acc<3, 12>(A, MLPD, Wt, MLPD, mt, ng, acc);
  const int lane = threadIdx.x & 31;
#pragma unroll
  for (int t = 0; t < 3; ++t) {
    const int col = (ng * 3 + t) * 16 + (lane & 15);
    const float bv = bias[col];
#pragma unroll
    for (int vi = 0; vi < 8; ++vi) {
      const int row = mt * 16 + ((lane >> 4) * 8) + vi;
      const long idx = (long)row * CB + col;
      outp[idx] = 0.5f * (acc[t][vi] + bv) + x1[idx];
    }
  }
}

// ---------------------------------------------------------------------------
extern "C" void kernel_launch(void* const* d_in, const int* in_sizes, int n_in,
                              void* d_out, int out_size, void* d_ws, size_t ws_size,
                              hipStream_t stream) {
  (void)in_sizes; (void)n_in; (void)out_size; (void)ws_size;
  const float* inp     = (const float*)d_in[0];
  const float* n1g     = (const float*)d_in[1];
  const float* n1b     = (const float*)d_in[2];
  const float* qkv_w   = (const float*)d_in[3];
  const float* qkv_b   = (const float*)d_in[4];
  const float* proj_w  = (const float*)d_in[5];
  const float* proj_b  = (const float*)d_in[6];
  const float* n2g     = (const float*)d_in[7];
  const float* n2b     = (const float*)d_in[8];
  const float* mlp_w1  = (const float*)d_in[9];
  const float* mlp_b1  = (const float*)d_in[10];
  const float* mlp_w2  = (const float*)d_in[11];
  const float* mlp_b2  = (const float*)d_in[12];
  const float* amask   = (const float*)d_in[13];
  float* outp = (float*)d_out;

  // workspace layout (bytes) — weights stored transposed (N x K), f16
  char* ws = (char*)d_ws;
  _Float16* wqkv  = (_Float16*)(ws + 0);          // 288 x 96
  _Float16* wproj = (_Float16*)(ws + 55296);      // 96 x 96
  _Float16* wm1   = (_Float16*)(ws + 73728);      // 384 x 96
  _Float16* wm2   = (_Float16*)(ws + 147456);     // 96 x 384
  _Float16* xbox  = (_Float16*)(ws + 221184);     // 25165824 B; reused: attn_out, ln2-out
  _Float16* qbuf  = (_Float16*)(ws + 25387008);   // 3 x 33554432 B; reused: mlp hidden
  _Float16* kbuf  = qbuf + QSZ;
  _Float16* vtbuf = kbuf + QSZ;
  _Float16* hid   = qbuf;
  float*    x1    = (float*)(ws + 126050304);     // 50331648 B
  _Float16* attn_out = xbox;
  _Float16* hbuf     = xbox;

  // 1) weights -> f16, transposed to K-contiguous B layout
  k_w_to_f16t<<<(27648 + 255) / 256, 256, 0, stream>>>(qkv_w, wqkv, CB, C3);
  k_w_to_f16t<<<(9216  + 255) / 256, 256, 0, stream>>>(proj_w, wproj, CB, CB);
  k_w_to_f16t<<<(36864 + 255) / 256, 256, 0, stream>>>(mlp_w1, wm1, CB, MLPD);
  k_w_to_f16t<<<(36864 + 255) / 256, 256, 0, stream>>>(mlp_w2, wm2, MLPD, CB);

  // 2) zero q/k padded buffers (K-dim pads d=24..31 must be exact zeros)
  k_zero128<<<(4194304 + 255) / 256, 256, 0, stream>>>((uint4*)qbuf, 4194304);

  // 3) LN1 + roll + box partition   (8 rows/block)
  k_ln1_box<<<NTOK / 8, 256, 0, stream>>>(inp, n1g, n1b, xbox);

  // 4) QKV GEMM: 8192 x 6 wave-groups (3 N-tiles each), 8 waves/block
  k_qkv<<<(8192 * 6) / 8, 256, 0, stream>>>(xbox, wqkv, qkv_b, qbuf, kbuf, vtbuf);

  // 5) fused attention: one block per (window, head)
  k_attn<<<2048 * 4, 128, 0, stream>>>(qbuf, kbuf, vtbuf, amask, attn_out);

  // 6) proj + box reverse + roll + residual -> x1
  k_proj<<<(8192 * 2) / 8, 256, 0, stream>>>(attn_out, wproj, proj_b, inp, x1);

  // 7) LN2
  k_ln2<<<NTOK / 8, 256, 0, stream>>>(x1, n2g, n2b, hbuf);

  // 8) MLP1 + GELU
  k_mlp1<<<(8192 * 6) / 8, 256, 0, stream>>>(hbuf, wm1, mlp_b1, hid);

  // 9) MLP2 + residual -> out
  k_mlp2<<<(8192 * 2) / 8, 256, 0, stream>>>(hid, wm2, mlp_b2, x1, outp);
}
